// STTransformerBlock_9380208575129
// MI455X (gfx1250) — matrix operand order repaired and checked by reading, not compiled, not run on hardware
//
#include <hip/hip_runtime.h>

// ---------------------------------------------------------------------------
// MI455X (gfx1250) ST-Transformer block.
// GEMMs: v_wmma_f32_16x16x32_bf16, A tiles staged with
// global_load_async_to_lds_b128 (ASYNCcnt) + double buffering; B tiles staged
// synchronously with on-the-fly fp32->bf16 convert (weights are read once, so
// converting them in-flight is the bandwidth-optimal choice on 23.3 TB/s HBM).
// Activations are kept bf16 end-to-end; fp32 only for residual/LN/softmax math.
// ---------------------------------------------------------------------------

typedef __attribute__((ext_vector_type(16))) __bf16 v16bf;
typedef __attribute__((ext_vector_type(8)))  float  v8f;

union FragBF { v16bf v; unsigned int u[8]; };

__device__ __forceinline__ unsigned short f2bf(float f) {
    union { float f; unsigned int u; } c; c.f = f;
    unsigned int u = c.u;
    u += 0x7fffu + ((u >> 16) & 1u);   // round-to-nearest-even
    return (unsigned short)(u >> 16);
}
__device__ __forceinline__ float bf2f(unsigned short u) {
    union { unsigned int u; float f; } c; c.u = ((unsigned int)u) << 16;
    return c.f;
}

#define BM 128
#define BN 128
#define BK 32
// LDS: A double-buffered, rows padded to 40 ushorts (80B, 16B-multiple, no
// 16-lane bank conflicts). B rows 34 ushorts (pair-contiguous K).
#define A_BUF_BYTES (BM * 40 * 2)              // 10240
#define B_BUF_BYTES (BN * 34 * 2)              // 8704
#define SMEM_A_BYTES (2 * A_BUF_BYTES)         // 20480
#define SMEM_BYTES (SMEM_A_BYTES + 2 * B_BUF_BYTES)   // 37888

// EPI: 0 -> C = alpha*acc + bias ; 1 -> C += rowscale[row]*(alpha*acc + bias)
// BTRANS: 0 -> B[k][n] = B[k*ldb+n] ; 1 -> B[k][n] = B[n*ldb+k]
// OUTBF: output bf16 (1) or fp32 (0).  BSRC: B global is fp32 (0) or bf16 (1).
template<int EPI, int BTRANS, int OUTBF, int BSRC>
__global__ __launch_bounds__(256) void gemm_bf16_kernel(
    const unsigned short* __restrict__ Ap, int lda, long aS1, long aS2,
    const void* __restrict__ Bvp, int ldb, long bS1, long bS2,
    void* __restrict__ Cvp, int ldc, long cS1, long cS2,
    const float* __restrict__ bias,
    const float* __restrict__ rowscale, int rsStride,
    float alpha, int M, int N, int K, int zdiv)
{
    extern __shared__ char smem[];

    const int z  = blockIdx.z;
    const int z1 = z / zdiv, z2 = z - z1 * zdiv;
    Ap += z1 * aS1 + z2 * aS2;
    const long boff = z1 * bS1 + z2 * bS2;
    const long coff = z1 * cS1 + z2 * cS2;

    const int tid   = threadIdx.x;
    const int lane  = tid & 31;
    const int wvid  = tid >> 5;
    const int wm    = wvid & 3;     // 4 waves along M
    const int wn    = wvid >> 2;    // 2 waves along N
    const int l15   = lane & 15;
    const int khalf = lane >> 4;
    const long m0   = (long)blockIdx.y * BM;
    const int  n0   = blockIdx.x * BN;

    v8f acc[2][4];
    for (int i = 0; i < 2; ++i)
        for (int j = 0; j < 4; ++j)
            for (int r = 0; r < 8; ++r) acc[i][j][r] = 0.0f;

    // ---- stage helpers ----
    auto stageA = [&](int buf, int kb) {   // async bf16 B128: 512B/wave/op
        unsigned abase = (unsigned)(buf * A_BUF_BYTES);
        #pragma unroll
        for (int i = 0; i < 2; ++i) {
            int chunk = tid + i * 256;          // 512 chunks of 8 bf16
            int kslot = chunk & 3, mm = chunk >> 2;
            unsigned lofs = abase + (unsigned)((mm * 40 + kslot * 8) * 2);
            const unsigned short* gp = Ap + (m0 + mm) * lda + kb + kslot * 8;
            asm volatile("global_load_async_to_lds_b128 %0, %1, off"
                         :: "v"(lofs), "v"(gp) : "memory");
        }
    };
    auto stageB = [&](int buf, int kb) {   // sync, converts fp32->bf16 if needed
        unsigned short* Bsm =
            (unsigned short*)(smem + SMEM_A_BYTES + buf * B_BUF_BYTES);
        #pragma unroll
        for (int i = 0; i < 16; ++i) {
            int idx = i * 256 + tid;
            int nn, kk;
            if (BTRANS) { kk = idx & 31;  nn = idx >> 5; }
            else        { nn = idx & 127; kk = idx >> 7; }
            int gn = n0 + nn;
            unsigned short hb = 0;
            if (gn < N) {
                long gi = boff + (BTRANS ? (long)gn * ldb + kb + kk
                                         : (long)(kb + kk) * ldb + gn);
                if (BSRC == 0) hb = f2bf(((const float*)Bvp)[gi]);
                else           hb = ((const unsigned short*)Bvp)[gi];
            }
            Bsm[nn * 34 + kk] = hb;
        }
    };

    // prologue: stage 0 into buffer 0
    stageA(0, 0);
    stageB(0, 0);

    int it = 0;
    for (int kb = 0; kb < K; kb += BK, ++it) {
        const int buf = it & 1;
        asm volatile("s_wait_asynccnt 0x0" ::: "memory");
        __syncthreads();                       // stage(it) visible to all waves
        if (kb + BK < K) {                     // overlap next stage with math
            stageA(buf ^ 1, kb + BK);
            stageB(buf ^ 1, kb + BK);
        }

        const unsigned short* Asm =
            (const unsigned short*)(smem + buf * A_BUF_BYTES);
        const unsigned short* Bsm =
            (const unsigned short*)(smem + SMEM_A_BYTES + buf * B_BUF_BYTES);

        FragBF afr[2], bfr[4];
        #pragma unroll
        for (int i = 0; i < 2; ++i) {
            int row = wm * 32 + i * 16 + l15;
            #pragma unroll
            for (int vv = 0; vv < 8; ++vv) {
                int kk = ((vv >> 2) << 4) + (khalf << 3) + ((vv & 3) << 1);
                afr[i].u[vv] = *(const unsigned int*)&Asm[row * 40 + kk];
            }
        }
        #pragma unroll
        for (int j = 0; j < 4; ++j) {
            int col = wn * 64 + j * 16 + l15;
            #pragma unroll
            for (int vv = 0; vv < 8; ++vv) {
                int kk = ((vv >> 2) << 4) + (khalf << 3) + ((vv & 3) << 1);
                bfr[j].u[vv] = *(const unsigned int*)&Bsm[col * 34 + kk];
            }
        }
        #pragma unroll
        for (int i = 0; i < 2; ++i)
            #pragma unroll
            for (int j = 0; j < 4; ++j)
                acc[i][j] = __builtin_amdgcn_wmma_f32_16x16x32_bf16(
                    false, afr[i].v, false, bfr[j].v,
                    (short)0, acc[i][j], false, false);
    }

    // ---- epilogue ----
    #pragma unroll
    for (int i = 0; i < 2; ++i) {
        #pragma unroll
        for (int j = 0; j < 4; ++j) {
            int col = n0 + wn * 64 + j * 16 + l15;
            if (col < N) {
                #pragma unroll
                for (int r = 0; r < 8; ++r) {
                    long row = m0 + wm * 32 + i * 16 + khalf * 8 + r;
                    float val = alpha * acc[i][j][r];
                    if (bias) val += bias[col];
                    long ci = coff + row * ldc + col;
                    if (OUTBF) {
                        ((unsigned short*)Cvp)[ci] = f2bf(val);
                    } else {
                        float* cp = &((float*)Cvp)[ci];
                        if (EPI == 1) *cp += rowscale[row * rsStride] * val;
                        else          *cp = val;
                    }
                }
            }
        }
    }
}

// ---------------------------------------------------------------------------
// fp32 -> bf16 elementwise convert
// ---------------------------------------------------------------------------
__global__ __launch_bounds__(256) void f32_to_bf16_kernel(
    const float* __restrict__ src, unsigned short* __restrict__ dst, long n)
{
    long i = (long)blockIdx.x * blockDim.x + threadIdx.x;
    if (i < n) dst[i] = f2bf(src[i]);
}

// ---------------------------------------------------------------------------
// Row softmax over 256 bf16 columns (spatial attention probs), in place.
// ---------------------------------------------------------------------------
__global__ __launch_bounds__(256) void softmax256_kernel(unsigned short* __restrict__ s)
{
    __shared__ float red[8];
    unsigned short* p = s + (long)blockIdx.x * 256;
    const int tid = threadIdx.x, lane = tid & 31, w = tid >> 5;

    float v = bf2f(p[tid]);
    float m = v;
    for (int off = 16; off > 0; off >>= 1) m = fmaxf(m, __shfl_xor(m, off));
    if (lane == 0) red[w] = m;
    __syncthreads();
    float mx = red[0];
    for (int i = 1; i < 8; ++i) mx = fmaxf(mx, red[i]);
    __syncthreads();

    float e = __expf(v - mx);
    float ssum = e;
    for (int off = 16; off > 0; off >>= 1) ssum += __shfl_xor(ssum, off);
    if (lane == 0) red[w] = ssum;
    __syncthreads();
    float tot = 0.0f;
    for (int i = 0; i < 8; ++i) tot += red[i];
    p[tid] = f2bf(e / tot);
}

// ---------------------------------------------------------------------------
// Temporal causal attention: T=8, D=64, bf16 q/k/v/o. Tiny FLOPs -> VALU.
// ---------------------------------------------------------------------------
__global__ __launch_bounds__(128) void temporal_attn_kernel(
    const unsigned short* __restrict__ q, const unsigned short* __restrict__ k,
    const unsigned short* __restrict__ v, unsigned short* __restrict__ o)
{
    int idx = blockIdx.x * blockDim.x + threadIdx.x;
    if (idx >= 2 * 256 * 12 * 8) return;
    int t = idx & 7;
    int h = (idx >> 3) % 12;
    int p = (idx / 96) & 255;
    int b = idx / (96 * 256);

    long qoff = ((long)((b * 8 + t) * 256 + p)) * 768 + h * 64;
    float qv[64];
    #pragma unroll
    for (int d = 0; d < 64; ++d) qv[d] = bf2f(q[qoff + d]);

    float sc[8];
    float mx = -1e30f;
    #pragma unroll
    for (int s = 0; s < 8; ++s) {
        float dot = -1e30f;
        if (s <= t) {
            long koff = ((long)((b * 8 + s) * 256 + p)) * 768 + h * 64;
            float a = 0.0f;
            #pragma unroll
            for (int d = 0; d < 64; ++d) a += qv[d] * bf2f(k[koff + d]);
            dot = a * 0.125f;
        }
        sc[s] = dot;
        mx = fmaxf(mx, dot);
    }
    float sum = 0.0f;
    #pragma unroll
    for (int s = 0; s < 8; ++s) {
        float e = (s <= t) ? __expf(sc[s] - mx) : 0.0f;
        sc[s] = e; sum += e;
    }
    float inv = 1.0f / sum;

    float out[64];
    #pragma unroll
    for (int d = 0; d < 64; ++d) out[d] = 0.0f;
    #pragma unroll
    for (int s = 0; s < 8; ++s) {
        if (s <= t) {
            float w = sc[s] * inv;
            long voff = ((long)((b * 8 + s) * 256 + p)) * 768 + h * 64;
            #pragma unroll
            for (int d = 0; d < 64; ++d) out[d] += w * bf2f(v[voff + d]);
        }
    }
    #pragma unroll
    for (int d = 0; d < 64; ++d) o[qoff + d] = f2bf(out[d]);
}

// ---------------------------------------------------------------------------
// y = LayerNorm(x + r)*g + b  (E=768). Writes fp32 y and optional bf16 copy.
// ---------------------------------------------------------------------------
__global__ __launch_bounds__(256) void add_ln_kernel(
    const float* __restrict__ x, const float* __restrict__ r,
    const float* __restrict__ g, const float* __restrict__ bb,
    float* __restrict__ y, unsigned short* __restrict__ yb)
{
    __shared__ float red0[8], red1[8];
    const long base = (long)blockIdx.x * 768;
    const int tid = threadIdx.x, lane = tid & 31, w = tid >> 5;

    float vals[3];
    float s = 0.0f, s2 = 0.0f;
    #pragma unroll
    for (int j = 0; j < 3; ++j) {
        int col = tid + j * 256;
        float v = x[base + col] + r[base + col];
        vals[j] = v; s += v; s2 += v * v;
    }
    for (int off = 16; off > 0; off >>= 1) {
        s  += __shfl_xor(s, off);
        s2 += __shfl_xor(s2, off);
    }
    if (lane == 0) { red0[w] = s; red1[w] = s2; }
    __syncthreads();
    float ts = 0.0f, ts2 = 0.0f;
    for (int i = 0; i < 8; ++i) { ts += red0[i]; ts2 += red1[i]; }
    float mean = ts * (1.0f / 768.0f);
    float var  = ts2 * (1.0f / 768.0f) - mean * mean;
    float rstd = rsqrtf(var + 1e-5f);
    #pragma unroll
    for (int j = 0; j < 3; ++j) {
        int col = tid + j * 256;
        float o = (vals[j] - mean) * rstd * g[col] + bb[col];
        y[base + col] = o;
        if (yb) yb[base + col] = f2bf(o);
    }
}

// ---------------------------------------------------------------------------
// Router: top-2 of 8 logits -> softmax over the two -> combine[N,8]
// ---------------------------------------------------------------------------
__global__ __launch_bounds__(256) void router_top2_kernel(
    const float* __restrict__ logits, float* __restrict__ combine)
{
    int n = blockIdx.x * blockDim.x + threadIdx.x;
    if (n >= 4096) return;
    float l[8];
    #pragma unroll
    for (int e = 0; e < 8; ++e) l[e] = logits[n * 8 + e];
    int i0 = 0; float b0 = l[0];
    #pragma unroll
    for (int e = 1; e < 8; ++e) if (l[e] > b0) { b0 = l[e]; i0 = e; }
    int i1 = -1; float b1 = -1e30f;
    #pragma unroll
    for (int e = 0; e < 8; ++e) if (e != i0 && l[e] > b1) { b1 = l[e]; i1 = e; }
    float w0 = 1.0f / (1.0f + __expf(b1 - b0));
    float w1 = 1.0f - w0;
    #pragma unroll
    for (int e = 0; e < 8; ++e) combine[n * 8 + e] = 0.0f;
    combine[n * 8 + i0] = w0;
    combine[n * 8 + i1] = w1;
}

// hidden = silu(hv) * hg  -> bf16
__global__ __launch_bounds__(256) void silu_gate_kernel(
    const float* __restrict__ hv, const float* __restrict__ hg,
    unsigned short* __restrict__ hid, long total)
{
    long i = (long)blockIdx.x * blockDim.x + threadIdx.x;
    if (i >= total) return;
    float a = hv[i];
    hid[i] = f2bf((a / (1.0f + __expf(-a))) * hg[i]);
}

// ---------------------------------------------------------------------------
// Launcher
// ---------------------------------------------------------------------------
extern "C" void kernel_launch(void* const* d_in, const int* in_sizes, int n_in,
                              void* d_out, int out_size, void* d_ws, size_t ws_size,
                              hipStream_t stream) {
    const float* x       = (const float*)d_in[0];
    const float* sa_wq   = (const float*)d_in[1];
    const float* sa_wk   = (const float*)d_in[2];
    const float* sa_wv   = (const float*)d_in[3];
    const float* sa_wo   = (const float*)d_in[4];
    const float* sa_bq   = (const float*)d_in[5];
    const float* sa_bk   = (const float*)d_in[6];
    const float* sa_bv   = (const float*)d_in[7];
    const float* sa_bo   = (const float*)d_in[8];
    const float* sa_g    = (const float*)d_in[9];
    const float* sa_bn   = (const float*)d_in[10];
    const float* ta_wq   = (const float*)d_in[11];
    const float* ta_wk   = (const float*)d_in[12];
    const float* ta_wv   = (const float*)d_in[13];
    const float* ta_wo   = (const float*)d_in[14];
    const float* ta_bq   = (const float*)d_in[15];
    const float* ta_bk   = (const float*)d_in[16];
    const float* ta_bv   = (const float*)d_in[17];
    const float* ta_bo   = (const float*)d_in[18];
    const float* ta_g    = (const float*)d_in[19];
    const float* ta_bn   = (const float*)d_in[20];
    const float* rw      = (const float*)d_in[21];
    const float* e_wv    = (const float*)d_in[22];
    const float* e_bv    = (const float*)d_in[23];
    const float* e_wg    = (const float*)d_in[24];
    const float* e_bg    = (const float*)d_in[25];
    const float* e_wo    = (const float*)d_in[26];
    const float* e_bo    = (const float*)d_in[27];
    const float* ffn_g   = (const float*)d_in[28];
    const float* ffn_bn  = (const float*)d_in[29];
    float* out = (float*)d_out;

    const long NE = (long)4096 * 768;        // 3,145,728
    const long NH = (long)4096 * 2048;       // 8,388,608

    char* W = (char*)d_ws;
    long off = 0;
    auto takeF = [&](long n) { float* p = (float*)(W + off); off += n * 4; return p; };
    auto takeH = [&](long n) { unsigned short* p = (unsigned short*)(W + off); off += n * 2; return p; };

    float*          x1f  = takeF(NE);
    float*          x2f  = takeF(NE);
    float*          proj = takeF(NE);
    float*          moe  = takeF(NE);
    float*          hvb  = takeF(NH);        // also hosts bf16 probs (aliased)
    float*          hgb  = takeF(NH);
    float*          rlog = takeF(4096 * 8);
    float*          comb = takeF(4096 * 8);
    unsigned short* xb   = takeH(NE);
    unsigned short* qb   = takeH(NE);
    unsigned short* kb   = takeH(NE);
    unsigned short* vb   = takeH(NE);
    unsigned short* ob   = takeH(NE);
    unsigned short* x1b  = takeH(NE);
    unsigned short* x2b  = takeH(NE);
    unsigned short* hid  = takeH(NH);
    unsigned short* probs = (unsigned short*)hvb;   // dead before MoE writes hv

    dim3 blk(256);
    auto G = [](int n, int m, int z) {
        return dim3((unsigned)((n + 127) / 128), (unsigned)((m + 127) / 128), (unsigned)z);
    };
    const long BTS = 256L * 768;             // per-(b,t) slab stride (elements)

    // input -> bf16
    f32_to_bf16_kernel<<<(unsigned)((NE + 255) / 256), blk, 0, stream>>>(x, xb, NE);

    // ================= spatial attention =================
    gemm_bf16_kernel<0,0,1,0><<<G(768,4096,1), blk, SMEM_BYTES, stream>>>(
        xb,768,0,0, sa_wq,768,0,0, qb,768,0,0, sa_bq, nullptr,0, 1.0f, 4096,768,768, 1);
    gemm_bf16_kernel<0,0,1,0><<<G(768,4096,1), blk, SMEM_BYTES, stream>>>(
        xb,768,0,0, sa_wk,768,0,0, kb,768,0,0, sa_bk, nullptr,0, 1.0f, 4096,768,768, 1);
    gemm_bf16_kernel<0,0,1,0><<<G(768,4096,1), blk, SMEM_BYTES, stream>>>(
        xb,768,0,0, sa_wv,768,0,0, vb,768,0,0, sa_bv, nullptr,0, 1.0f, 4096,768,768, 1);

    // S = (Q Kt)/8, z = bt*12 + h ; bf16 out, softmaxed in place
    gemm_bf16_kernel<0,1,1,1><<<G(256,256,192), blk, SMEM_BYTES, stream>>>(
        qb,768, BTS,64, kb,768, BTS,64,
        probs,256, 12L*65536,65536, nullptr, nullptr,0, 0.125f, 256,256,64, 12);
    softmax256_kernel<<<192 * 256, blk, 0, stream>>>(probs);
    // O = A V
    gemm_bf16_kernel<0,0,1,1><<<G(64,256,192), blk, SMEM_BYTES, stream>>>(
        probs,256, 12L*65536,65536, vb,768, BTS,64,
        ob,768, BTS,64, nullptr, nullptr,0, 1.0f, 256,64,256, 12);

    gemm_bf16_kernel<0,0,0,0><<<G(768,4096,1), blk, SMEM_BYTES, stream>>>(
        ob,768,0,0, sa_wo,768,0,0, proj,768,0,0, sa_bo, nullptr,0, 1.0f, 4096,768,768, 1);
    add_ln_kernel<<<4096, blk, 0, stream>>>(x, proj, sa_g, sa_bn, x1f, x1b);

    // ================= temporal attention =================
    gemm_bf16_kernel<0,0,1,0><<<G(768,4096,1), blk, SMEM_BYTES, stream>>>(
        x1b,768,0,0, ta_wq,768,0,0, qb,768,0,0, ta_bq, nullptr,0, 1.0f, 4096,768,768, 1);
    gemm_bf16_kernel<0,0,1,0><<<G(768,4096,1), blk, SMEM_BYTES, stream>>>(
        x1b,768,0,0, ta_wk,768,0,0, kb,768,0,0, ta_bk, nullptr,0, 1.0f, 4096,768,768, 1);
    gemm_bf16_kernel<0,0,1,0><<<G(768,4096,1), blk, SMEM_BYTES, stream>>>(
        x1b,768,0,0, ta_wv,768,0,0, vb,768,0,0, ta_bv, nullptr,0, 1.0f, 4096,768,768, 1);
    temporal_attn_kernel<<<(2*256*12*8 + 127) / 128, dim3(128), 0, stream>>>(qb, kb, vb, ob);
    gemm_bf16_kernel<0,0,0,0><<<G(768,4096,1), blk, SMEM_BYTES, stream>>>(
        ob,768,0,0, ta_wo,768,0,0, proj,768,0,0, ta_bo, nullptr,0, 1.0f, 4096,768,768, 1);
    add_ln_kernel<<<4096, blk, 0, stream>>>(x1f, proj, ta_g, ta_bn, x2f, x2b);

    // ================= MoE SwiGLU, top-2 =================
    gemm_bf16_kernel<0,0,0,0><<<G(8,4096,1), blk, SMEM_BYTES, stream>>>(
        x2b,768,0,0, rw,8,0,0, rlog,8,0,0, nullptr, nullptr,0, 1.0f, 4096,8,768, 1);
    router_top2_kernel<<<16, blk, 0, stream>>>(rlog, comb);
    hipMemsetAsync(moe, 0, NE * sizeof(float), stream);

    for (int e = 0; e < 8; ++e) {
        const float* wve = e_wv + (long)e * 768 * 2048;
        const float* wge = e_wg + (long)e * 768 * 2048;
        const float* woe = e_wo + (long)e * 2048 * 768;
        const float* bve = e_bv + (long)e * 2048;
        const float* bge = e_bg + (long)e * 2048;
        const float* boe = e_bo + (long)e * 768;
        gemm_bf16_kernel<0,0,0,0><<<G(2048,4096,1), blk, SMEM_BYTES, stream>>>(
            x2b,768,0,0, wve,2048,0,0, hvb,2048,0,0, bve, nullptr,0, 1.0f, 4096,2048,768, 1);
        gemm_bf16_kernel<0,0,0,0><<<G(2048,4096,1), blk, SMEM_BYTES, stream>>>(
            x2b,768,0,0, wge,2048,0,0, hgb,2048,0,0, bge, nullptr,0, 1.0f, 4096,2048,768, 1);
        silu_gate_kernel<<<(unsigned)((NH + 255) / 256), blk, 0, stream>>>(hvb, hgb, hid, NH);
        gemm_bf16_kernel<1,0,0,0><<<G(768,4096,1), blk, SMEM_BYTES, stream>>>(
            hid,2048,0,0, woe,768,0,0, moe,768,0,0, boe, comb + e, 8, 1.0f, 4096,768,2048, 1);
    }

    add_ln_kernel<<<4096, blk, 0, stream>>>(x2f, moe, ffn_g, ffn_bn, out, nullptr);
}
